// EdgeTransformer_81844896792600
// MI455X (gfx1250) — matrix operand order repaired and checked by reading, compiled, not run on hardware
//
#include <hip/hip_runtime.h>

typedef __attribute__((ext_vector_type(16))) __bf16 v16bf;
typedef __attribute__((ext_vector_type(8)))  float  v8f;

#define D_NODE 64
#define D_EDGE 32
#define D_IN   160
#define D_HID  320
#define D_OUT  32

#define KC1 5            // D_IN / 32   (K-chunks GEMM1)
#define NT1 20           // D_HID / 16  (N-tiles  GEMM1)
#define NT_BLK 5         // N-tiles kept live at once (40 acc VGPRs)
#define NBLK (NT1 / NT_BLK)   // 4
#define KC2 10           // D_HID / 32  (K-chunks GEMM2)
#define NT2 2            // D_OUT / 16  (N-tiles  GEMM2)
#define FRAG 512         // 32 lanes * 16 bf16 halves per fragment

#define W1P_ELEMS (KC1 * NT1 * FRAG)   // 51200 bf16
#define W2P_ELEMS (KC2 * NT2 * FRAG)   // 10240 bf16

#define NWAVES 8
#define TILE_EDGES 128                 // NWAVES * 16
#define TILES_PER_BLOCK 5
#define N_EDGE_TOT 1600000
#define NBLOCKS (N_EDGE_TOT / (TILE_EDGES * TILES_PER_BLOCK))  // 2500

#define SMEM_BF16 (W1P_ELEMS + W2P_ELEMS + NWAVES * KC1 * FRAG + NWAVES * KC2 * FRAG)
#define SMEM_BYTES (SMEM_BF16 * 2)     // 245760 B <= 320KB WGP LDS

// Native bf16 truncation: lowers to the hardware f32->bf16 convert on gfx1250.
__device__ __forceinline__ __bf16 tobf(float f) { return (__bf16)f; }

// Fast sigmoid: v_exp + v_rcp (1 TRANS each) instead of a full f32 divide.
__device__ __forceinline__ float fast_sigmoid(float x) {
  return __builtin_amdgcn_rcpf(1.0f + __expf(-x));
}

// Pack W1[160,320] and W2[320,32] (f32 row-major) into bf16 fragments in the
// wave32 WMMA B-operand layout: lane l -> N = nt*16 + (l&15), K-half index h
// -> K = kc*32 + h + (h&8) + ((l&16)>>1). Fragment order: kc-major, nt-minor.
__global__ void EdgeTf_pack_weights(const float* __restrict__ W1,
                                    const float* __restrict__ W2,
                                    __bf16* __restrict__ w1p,
                                    __bf16* __restrict__ w2p) {
  int i = blockIdx.x * blockDim.x + threadIdx.x;
  if (i < W1P_ELEMS) {
    int h = i & 15;
    int l = (i >> 4) & 31;
    int frag = i >> 9;
    int nt = frag % NT1;
    int kc = frag / NT1;
    int k = kc * 32 + h + (h & 8) + ((l & 16) >> 1);
    int n = nt * 16 + (l & 15);
    w1p[i] = tobf(W1[k * D_HID + n]);
  }
  int j = i - W1P_ELEMS;
  if (j >= 0 && j < W2P_ELEMS) {
    int h = j & 15;
    int l = (j >> 4) & 31;
    int frag = j >> 9;
    int nt = frag % NT2;
    int kc = frag / NT2;
    int k = kc * 32 + h + (h & 8) + ((l & 16) >> 1);
    int n = nt * 16 + (l & 15);
    w2p[j] = tobf(W2[k * D_OUT + n]);
  }
}

__global__ __launch_bounds__(256, 1) void EdgeTf_mlp_kernel(
    const float* __restrict__ nodes, const float* __restrict__ edges,
    const int* __restrict__ senders, const int* __restrict__ receivers,
    const float* __restrict__ b1, const float* __restrict__ b2,
    const __bf16* __restrict__ w1p, const __bf16* __restrict__ w2p,
    float* __restrict__ out)
{
  extern __shared__ __attribute__((aligned(32))) char smem_raw[];
  __bf16* w1s = (__bf16*)smem_raw;                 // 100 KB
  __bf16* w2s = w1s + W1P_ELEMS;                   //  20 KB
  __bf16* fps = w2s + W2P_ELEMS;                   //  40 KB (A-frags, feats)
  __bf16* hps = fps + NWAVES * KC1 * FRAG;         //  80 KB (A-frags, hidden)

  const int tid  = threadIdx.x;
  const int lane = tid & 31;
  const int wave = tid >> 5;

  // Stage the pre-packed bf16 weights into LDS once per block.
  {
    const uint4* s1 = (const uint4*)w1p;
    uint4* d1 = (uint4*)w1s;
    for (int i = tid; i < W1P_ELEMS * 2 / 16; i += 256) d1[i] = s1[i];
    const uint4* s2 = (const uint4*)w2p;
    uint4* d2 = (uint4*)w2s;
    for (int i = tid; i < W2P_ELEMS * 2 / 16; i += 256) d2[i] = s2[i];
  }

  for (int t = 0; t < TILES_PER_BLOCK; ++t) {
    const long long edgeBase =
        ((long long)blockIdx.x * TILES_PER_BLOCK + t) * TILE_EDGES;

    __syncthreads();   // weights visible (t==0); fps safe to reuse (t>0)

    // Fused gather: build the 128x160 feature tile (sender||receiver||edge)
    // as bf16, written straight into the wave32 WMMA A-operand layout.
    // A 16x32 bf16 layout: lane l holds row M=l&15; halves h map to
    // K = h + (h&8) + ((l&16)>>1) within the 32-wide K-chunk.
    for (int i = tid; i < NWAVES * KC1 * FRAG; i += 256) {
      int h  = i & 15;
      int l  = (i >> 4) & 31;
      int kc = (i >> 9) % KC1;
      int w  = i / (KC1 * FRAG);
      int m  = l & 15;
      int k  = kc * 32 + h + (h & 8) + ((l & 16) >> 1);
      long long e = edgeBase + w * 16 + m;
      float v;
      if (k < D_NODE) {
        v = nodes[(long long)senders[e] * D_NODE + k];
      } else if (k < 2 * D_NODE) {
        v = nodes[(long long)receivers[e] * D_NODE + (k - D_NODE)];
      } else {
        v = edges[e * D_EDGE + (k - 2 * D_NODE)];
      }
      fps[i] = tobf(v);
    }
    __syncthreads();

    // ---- GEMM1: h = sigmoid(feats @ W1 + b1); per-wave 16x320 tile,
    // processed in N-blocks of NT_BLK*16 columns to cap register pressure
    // at 5 live accumulators (40 VGPRs). ----
    const v16bf* afr = (const v16bf*)(fps + wave * KC1 * FRAG);
    const v16bf* bfr = (const v16bf*)w1s;
    __bf16* hw = hps + wave * KC2 * FRAG;

    #pragma unroll
    for (int nb = 0; nb < NBLK; ++nb) {
      v8f acc[NT_BLK];
      #pragma unroll
      for (int j = 0; j < NT_BLK; ++j) {
        float bias = b1[(nb * NT_BLK + j) * 16 + (lane & 15)];
        #pragma unroll
        for (int r = 0; r < 8; ++r) acc[j][r] = bias;
      }
      #pragma unroll
      for (int kc = 0; kc < KC1; ++kc) {
        v16bf a = afr[kc * 32 + lane];
        #pragma unroll
        for (int j = 0; j < NT_BLK; ++j) {
          v16bf b = bfr[(kc * NT1 + nb * NT_BLK + j) * 32 + lane];
          acc[j] = __builtin_amdgcn_wmma_f32_16x16x32_bf16(
              false, a, false, b, (short)0, acc[j], false, false);
        }
      }
      // Sigmoid this N-slice, scatter into the GEMM2 A-operand layout
      // (C-layout -> A-layout lane transpose through LDS), free the regs.
      #pragma unroll
      for (int j = 0; j < NT_BLK; ++j) {
        int k2  = (nb * NT_BLK + j) * 16 + (lane & 15);  // K index of GEMM2
        int kc2 = k2 >> 5;
        int kk  = k2 & 31;
        int lhi = ((kk >> 3) & 1) << 4;                  // high-lane select
        int h2  = (kk & 7) + (((kk >> 4) & 1) << 3);
        #pragma unroll
        for (int r = 0; r < 8; ++r) {
          int m = r + ((lane >> 4) << 3);                // C/D layout row
          hw[(kc2 * 32 + lhi + m) * 16 + h2] = tobf(fast_sigmoid(acc[j][r]));
        }
      }
    }
    __syncthreads();

    // ---- GEMM2: out = h @ W2 + b2 ----
    v8f acc2[NT2];
    #pragma unroll
    for (int nt = 0; nt < NT2; ++nt) {
      float bias = b2[nt * 16 + (lane & 15)];
      #pragma unroll
      for (int r = 0; r < 8; ++r) acc2[nt][r] = bias;
    }
    const v16bf* a2fr = (const v16bf*)hw;
    const v16bf* b2fr = (const v16bf*)w2s;
    #pragma unroll
    for (int kc = 0; kc < KC2; ++kc) {
      v16bf a = a2fr[kc * 32 + lane];
      #pragma unroll
      for (int nt = 0; nt < NT2; ++nt) {
        v16bf b = b2fr[(kc * NT2 + nt) * 32 + lane];
        acc2[nt] = __builtin_amdgcn_wmma_f32_16x16x32_bf16(
            false, a, false, b, (short)0, acc2[nt], false, false);
      }
    }

    // Store f32 output (16 edges x 32 cols per wave).
    const long long myBase = edgeBase + wave * 16;
    #pragma unroll
    for (int nt = 0; nt < NT2; ++nt) {
      int n = nt * 16 + (lane & 15);
      #pragma unroll
      for (int r = 0; r < 8; ++r) {
        int m = r + ((lane >> 4) << 3);
        out[(myBase + m) * D_OUT + n] = acc2[nt][r];
      }
    }
  }
}

extern "C" void kernel_launch(void* const* d_in, const int* in_sizes, int n_in,
                              void* d_out, int out_size, void* d_ws, size_t ws_size,
                              hipStream_t stream) {
  (void)in_sizes; (void)n_in; (void)out_size; (void)ws_size;
  const float* nodes     = (const float*)d_in[0];
  const float* edges     = (const float*)d_in[1];
  const int*   senders   = (const int*)d_in[2];
  const int*   receivers = (const int*)d_in[3];
  const float* W1        = (const float*)d_in[4];
  const float* b1        = (const float*)d_in[5];
  const float* W2        = (const float*)d_in[6];
  const float* b2        = (const float*)d_in[7];
  float* out = (float*)d_out;

  __bf16* w1p = (__bf16*)d_ws;                  // 51200 bf16
  __bf16* w2p = w1p + W1P_ELEMS;                // 10240 bf16 (ws >= 123 KB)

  EdgeTf_pack_weights<<<(W1P_ELEMS + W2P_ELEMS) / 256, 256, 0, stream>>>(
      W1, W2, w1p, w2p);
  EdgeTf_mlp_kernel<<<NBLOCKS, 256, SMEM_BYTES, stream>>>(
      nodes, edges, senders, receivers, b1, b2, w1p, w2p, out);
}
